// MultiHeadAttn_55946243998184
// MI455X (gfx1250) — matrix operand order repaired and checked by reading, not compiled, not run on hardware
//
#include <hip/hip_runtime.h>

// MultiHeadAttn (banded) for MI455X / gfx1250: bf16 WMMA everywhere, fp32 softmax,
// TDM (tensor_load_to_lds) staging of V key-tiles in the attention inner loop.
// B=2, T=4096, D=512, H=8, S=64, CONTEXT=256.

#define TLEN   4096
#define DMODEL 512
#define NHEADS 8
#define HSZ    64
#define CTX    256
#define BATCH  2

typedef __bf16 bf16_t;
typedef __attribute__((ext_vector_type(16))) __bf16 v16bf;
typedef __attribute__((ext_vector_type(8)))  __bf16 v8bf;
typedef __attribute__((ext_vector_type(8)))  float  v8f;

typedef __attribute__((ext_vector_type(4))) unsigned int u32x4;
typedef __attribute__((ext_vector_type(8))) int          i32x8;
typedef __attribute__((ext_vector_type(4))) int          i32x4;

#if defined(__HIP_DEVICE_COMPILE__) && defined(__has_builtin)
# if __has_builtin(__builtin_amdgcn_tensor_load_to_lds) && __has_builtin(__builtin_amdgcn_s_wait_tensorcnt)
#  define HAVE_TDM 1
# endif
#endif

static __device__ __forceinline__ v8f wmma_bf16(v16bf a, v16bf b, v8f c) {
  // D = A(16x32 bf16) * B(32x16 bf16) + C(16x16 f32)
  return __builtin_amdgcn_wmma_f32_16x16x32_bf16(false, a, false, b, (short)0, c,
                                                 false, false);
}

// A fragment: 16x32 tile of a row-major [*, ld] bf16 matrix starting at `base`.
// A layout: lane%16 = row M; lane/16 selects K half (0 or 16);
// vector elems 0..15 = K kg..kg+15 (contiguous, matching B fragment).
static __device__ __forceinline__ v16bf load_afrag(const bf16_t* base, int ld, int lane) {
  const int m  = lane & 15;
  const int kg = (lane >> 4) << 4;  // 0 or 16
  const bf16_t* p = base + m * ld + kg;
  v8bf lo = *(const v8bf*)(p);
  v8bf hi = *(const v8bf*)(p + 8);
  v16bf a;
#pragma unroll
  for (int i = 0; i < 8; ++i) { a[i] = lo[i]; a[i + 8] = hi[i]; }
  return a;
}

// B fragment: 32x16 (KxN). baseT points at row n0 of the *transposed* (NxK,
// row-major, leading dim ld) matrix. B layout used here (same k order as A): lane%16 = col N,
// lane/16 selects K half (0 or 16); 16 contiguous K values per lane = 32B load.
static __device__ __forceinline__ v16bf load_bfrag(const bf16_t* baseT, int ld, int lane) {
  const int n  = lane & 15;
  const int kh = (lane >> 4) << 4;  // 0 or 16
  return *(const v16bf*)(baseT + n * ld + kh);
}

#if defined(HAVE_TDM)
// TDM: async DMA of a 32-row x 128-byte (32x64 bf16) tile, global -> LDS.
// D# per CDNA5 ISA ch.8: data_size=3 (8B units) => row = 16 units.
static __device__ __forceinline__ void tdm_load_v(unsigned lds_addr, const bf16_t* gptr,
                                                  int rows_avail) {
  unsigned long long ga = (unsigned long long)(size_t)gptr;
  u32x4 g0 = { 1u,                                  // count=1, is_restore=0, gather=0
               lds_addr,                            // lds_addr (bytes)
               (unsigned)ga,                        // global_addr[31:0]
               ((unsigned)(ga >> 32) & 0x01FFFFFFu) | 0x80000000u }; // addr[56:32] | type=2
  i32x8 g1 = { (int)0x00030000u,                          // wg_mask=0, data_size=3 (8B)
               (int)(16u << 16),                          // tensor_dim0[15:0]=16 units/row
               (int)((unsigned)rows_avail << 16),         // tensor_dim1[15:0]=rows_avail
               (int)((16u << 16) | ((unsigned)rows_avail >> 16)), // tile_dim0=16 | dim1 hi
               32,                                        // tile_dim1=32 rows, tile_dim2=0
               16,                                        // tensor_dim0_stride=16 units
               0, 0 };
  i32x4 z4 = {0, 0, 0, 0};
#if __clang_major__ >= 23
  i32x8 z8 = {0, 0, 0, 0, 0, 0, 0, 0};
  __builtin_amdgcn_tensor_load_to_lds(g0, g1, z4, z4, z8, 0);
#else
  __builtin_amdgcn_tensor_load_to_lds(g0, g1, z4, z4, 0);
#endif
}
#endif

// ---------------- prep kernels ----------------

__global__ void cvt_f32_bf16(const float* __restrict__ x, bf16_t* __restrict__ y, int n) {
  int i = blockIdx.x * blockDim.x + threadIdx.x;
  if (i < n) y[i] = (bf16_t)x[i];
}

// w: [batch, R, C] f32  ->  wt: [batch, C, R] bf16
__global__ void transpose_to_bf16(const float* __restrict__ w, bf16_t* __restrict__ wt,
                                  int R, int C, int n) {
  int i = blockIdx.x * blockDim.x + threadIdx.x;
  if (i >= n) return;
  int rc  = R * C;
  int b   = i / rc;
  int rem = i - b * rc;
  int r   = rem / C;
  int c   = rem - r * C;
  wt[(size_t)b * rc + (size_t)c * R + r] = (bf16_t)w[i];
}

// ---------------- Q/V projections ----------------
// One wave -> one 16-row tile of Q[b,h] (16x64) and V[b,h] (16x64). K = 512.
__global__ __launch_bounds__(128) void proj_kernel(const bf16_t* __restrict__ Xbf,
                                                   const bf16_t* __restrict__ WQT,
                                                   const bf16_t* __restrict__ WVT,
                                                   bf16_t* __restrict__ Qbf,
                                                   bf16_t* __restrict__ Vbf,
                                                   bf16_t* __restrict__ VTbf) {
  const int lane = threadIdx.x & 31;
  const int wave = threadIdx.x >> 5;
  const int w    = blockIdx.x * 4 + wave;           // 0..4095
  const int t0   = (w & 255) << 4;
  const int hh   = (w >> 8) & 7;
  const int b    = w >> 11;

  const bf16_t* Xb = Xbf + ((size_t)b * TLEN + t0) * DMODEL;
  const bf16_t* Wq = WQT + (size_t)hh * HSZ * DMODEL;
  const bf16_t* Wv = WVT + (size_t)hh * HSZ * DMODEL;

  const v8f vz = {0.f, 0.f, 0.f, 0.f, 0.f, 0.f, 0.f, 0.f};
  v8f q[4] = {vz, vz, vz, vz};
  v8f v[4] = {vz, vz, vz, vz};

  for (int k0 = 0; k0 < DMODEL; k0 += 32) {
    v16bf a = load_afrag(Xb + k0, DMODEL, lane);
#pragma unroll
    for (int f = 0; f < 4; ++f) {
      q[f] = wmma_bf16(a, load_bfrag(Wq + (size_t)(f * 16) * DMODEL + k0, DMODEL, lane), q[f]);
      v[f] = wmma_bf16(a, load_bfrag(Wv + (size_t)(f * 16) * DMODEL + k0, DMODEL, lane), v[f]);
    }
  }

  const int half = lane >> 4, n = lane & 15;
  const size_t bh = (size_t)(b * NHEADS + hh);
  bf16_t* Qo  = Qbf  + (bh * TLEN + t0) * HSZ;
  bf16_t* Vo  = Vbf  + (bh * TLEN + t0) * HSZ;
  bf16_t* VTo = VTbf + bh * HSZ * TLEN;
#pragma unroll
  for (int f = 0; f < 4; ++f) {
#pragma unroll
    for (int r = 0; r < 8; ++r) {
      int row = 8 * half + r;
      int col = f * 16 + n;
      Qo[row * HSZ + col] = (bf16_t)q[f][r];
      float vv = v[f][r];
      Vo[row * HSZ + col] = (bf16_t)vv;
      VTo[(size_t)col * TLEN + t0 + row] = (bf16_t)vv;
    }
  }
}

// ---------------- banded flash attention ----------------
// One wave -> 16 query rows. Streams 32-wide key tiles over the ±256 band.
// V score-tiles are staged global->LDS by the Tensor Data Mover, overlapped
// with softmax/PV of the previous tile via TENSORcnt.
__global__ __launch_bounds__(128) void attn_kernel(const bf16_t* __restrict__ Qbf,
                                                   const bf16_t* __restrict__ Vbf,
                                                   const bf16_t* __restrict__ VTbf,
                                                   bf16_t* __restrict__ Cbf) {
  __shared__ __align__(16) bf16_t lds_p[4][16 * 32];
#if defined(HAVE_TDM)
  __shared__ __align__(16) bf16_t lds_v[4][32 * 64];
#endif
  const int lane = threadIdx.x & 31;
  // make per-wave scalars provably uniform (keeps TDM descriptor ops in SGPRs)
  const int w    = __builtin_amdgcn_readfirstlane(blockIdx.x * 4 + (threadIdx.x >> 5));
  const int wvu  = w & 3;                           // wave slot in block
  const int t0   = (w & 255) << 4;
  const int hh   = (w >> 8) & 7;
  const int b    = w >> 11;
  const int half = lane >> 4, n = lane & 15;

  const size_t bh   = (size_t)(b * NHEADS + hh);
  const bf16_t* Qb  = Qbf  + (bh * TLEN + t0) * HSZ;
  const bf16_t* Vb  = Vbf  + bh * TLEN * HSZ;
  const bf16_t* VTb = VTbf + bh * HSZ * TLEN;
  bf16_t* ldsw = &lds_p[wvu][0];

  // persistent Q fragments (16x64 = two K=32 A-frags)
  v16bf q0 = load_afrag(Qb, HSZ, lane);
  v16bf q1 = load_afrag(Qb + 32, HSZ, lane);

  const v8f vz = {0.f, 0.f, 0.f, 0.f, 0.f, 0.f, 0.f, 0.f};
  v8f O0 = vz, O1 = vz, O2 = vz, O3 = vz;     // 16x64 accumulator
  float mrow[8], lrow[8];
#pragma unroll
  for (int r = 0; r < 8; ++r) { mrow[r] = -3.0e38f; lrow[r] = 0.f; }

  int us = t0 - CTX; if (us < 0) us = 0; us &= ~31;
  int ue = t0 + 16 + CTX; if (ue > TLEN) ue = TLEN; ue = (ue + 31) & ~31;

#if defined(HAVE_TDM)
  bf16_t* vstage = &lds_v[wvu][0];
  const unsigned vlds = (unsigned)(size_t)vstage;   // low 32 bits = LDS byte address
  tdm_load_v(vlds, Vb + (size_t)us * HSZ, TLEN - us);
#endif

  for (int u0 = us; u0 < ue; u0 += 32) {
    // ---- fetch 32x64 V key tile (scores use B^T of V^T, i.e. V itself) ----
#if defined(HAVE_TDM)
    __builtin_amdgcn_s_wait_tensorcnt(0);
    asm volatile("" ::: "memory");
    const bf16_t* Va = vstage;
    v16bf ba0 = load_bfrag(Va, HSZ, lane);                  // cols u0..u0+15, K 0..31
    v16bf ba1 = load_bfrag(Va + 32, HSZ, lane);             // K 32..63
    v16bf bb0 = load_bfrag(Va + 16 * HSZ, HSZ, lane);       // cols u0+16..u0+31
    v16bf bb1 = load_bfrag(Va + 16 * HSZ + 32, HSZ, lane);
    // frags must be in VGPRs (and LDS free) before the next DMA overwrites it
    asm volatile("s_wait_dscnt 0" ::: "memory");
    if (u0 + 32 < ue)
      tdm_load_v(vlds, Vb + (size_t)(u0 + 32) * HSZ, TLEN - (u0 + 32));
#else
    const bf16_t* Va = Vb + (size_t)u0 * HSZ;
    v16bf ba0 = load_bfrag(Va, HSZ, lane);
    v16bf ba1 = load_bfrag(Va + 32, HSZ, lane);
    v16bf bb0 = load_bfrag(Va + 16 * HSZ, HSZ, lane);
    v16bf bb1 = load_bfrag(Va + 16 * HSZ + 32, HSZ, lane);
#endif
    // prefetch next VT tile (lowers to global_prefetch_b8)
    if (u0 + 32 < ue)
      __builtin_prefetch(VTb + u0 + 32 + (size_t)(half * 16 + n) * TLEN, 0, 1);

    v8f Sa = wmma_bf16(q1, ba1, wmma_bf16(q0, ba0, vz));
    v8f Sb = wmma_bf16(q1, bb1, wmma_bf16(q0, bb0, vz));

    // ---- scale + additive band mask (matches reference: -1e9 outside band) ----
    const int ua = u0 + n, ub = u0 + 16 + n;
    float pa[8], pb[8], rmax[8];
#pragma unroll
    for (int r = 0; r < 8; ++r) {
      int t = t0 + 8 * half + r;
      float sa = Sa[r] * 0.125f;
      float sb = Sb[r] * 0.125f;
      int da = t - ua; da = da < 0 ? -da : da;
      int db = t - ub; db = db < 0 ? -db : db;
      if (da > CTX) sa -= 1e9f;
      if (db > CTX) sb -= 1e9f;
      pa[r] = sa; pb[r] = sb;
      rmax[r] = fmaxf(sa, sb);
    }
    // row max across the 16 lanes of each half-wave (rows live across lanes)
#pragma unroll
    for (int r = 0; r < 8; ++r) {
      float vmx = rmax[r];
      vmx = fmaxf(vmx, __shfl_xor(vmx, 1));
      vmx = fmaxf(vmx, __shfl_xor(vmx, 2));
      vmx = fmaxf(vmx, __shfl_xor(vmx, 4));
      vmx = fmaxf(vmx, __shfl_xor(vmx, 8));
      rmax[r] = vmx;
    }
    float alpha[8];
#pragma unroll
    for (int r = 0; r < 8; ++r) {
      float mnew = fmaxf(mrow[r], rmax[r]);
      alpha[r]   = __expf(mrow[r] - mnew);
      mrow[r]    = mnew;
      float ea = __expf(pa[r] - mnew);
      float eb = __expf(pb[r] - mnew);
      pa[r] = ea; pb[r] = eb;
      float rs = ea + eb;
      rs += __shfl_xor(rs, 1);
      rs += __shfl_xor(rs, 2);
      rs += __shfl_xor(rs, 4);
      rs += __shfl_xor(rs, 8);
      lrow[r] = lrow[r] * alpha[r] + rs;
      O0[r] *= alpha[r]; O1[r] *= alpha[r]; O2[r] *= alpha[r]; O3[r] *= alpha[r];
    }

    // ---- D-frag -> A-frag layout change through per-wave LDS tile ----
#pragma unroll
    for (int r = 0; r < 8; ++r) {
      int row = 8 * half + r;
      ldsw[row * 32 + n]      = (bf16_t)pa[r];
      ldsw[row * 32 + 16 + n] = (bf16_t)pb[r];
    }
    __builtin_amdgcn_wave_barrier();           // DS ops are in-order per wave
    v16bf pA = load_afrag(ldsw, 32, lane);
    __builtin_amdgcn_wave_barrier();

    // ---- O += P(16x32) * V(32x64)  (B^T of V is VT, contiguous per lane) ----
    const bf16_t* VTu = VTb + u0;
    O0 = wmma_bf16(pA, load_bfrag(VTu + (size_t)0  * TLEN, TLEN, lane), O0);
    O1 = wmma_bf16(pA, load_bfrag(VTu + (size_t)16 * TLEN, TLEN, lane), O1);
    O2 = wmma_bf16(pA, load_bfrag(VTu + (size_t)32 * TLEN, TLEN, lane), O2);
    O3 = wmma_bf16(pA, load_bfrag(VTu + (size_t)48 * TLEN, TLEN, lane), O3);
  }

  // ---- normalize and emit concat-head layout [B, T, H*S] in bf16 ----
  float inv[8];
#pragma unroll
  for (int r = 0; r < 8; ++r) inv[r] = 1.0f / lrow[r];
  bf16_t* Cb = Cbf + ((size_t)b * TLEN + t0) * (NHEADS * HSZ) + hh * HSZ;
#pragma unroll
  for (int r = 0; r < 8; ++r) {
    int row = 8 * half + r;
    Cb[row * (NHEADS * HSZ) + 0  + n] = (bf16_t)(O0[r] * inv[r]);
    Cb[row * (NHEADS * HSZ) + 16 + n] = (bf16_t)(O1[r] * inv[r]);
    Cb[row * (NHEADS * HSZ) + 32 + n] = (bf16_t)(O2[r] * inv[r]);
    Cb[row * (NHEADS * HSZ) + 48 + n] = (bf16_t)(O3[r] * inv[r]);
  }
}

// ---------------- output projection: out = C @ WU ----------------
__global__ __launch_bounds__(128) void outproj_kernel(const bf16_t* __restrict__ Cbf,
                                                      const bf16_t* __restrict__ WUT,
                                                      float* __restrict__ out) {
  const int lane = threadIdx.x & 31;
  const int wave = threadIdx.x >> 5;
  const int w    = blockIdx.x * 4 + wave;    // 4096 waves
  const int colTile = w & 7;                 // 8 x 64-wide tiles over N=512
  const int rowTile = w >> 3;                // 512 x 16-row tiles over M=8192
  const bf16_t* Cb = Cbf + (size_t)rowTile * 16 * DMODEL;
  const int n0base = colTile * 64;

  const v8f vz = {0.f, 0.f, 0.f, 0.f, 0.f, 0.f, 0.f, 0.f};
  v8f acc[4] = {vz, vz, vz, vz};
  for (int k0 = 0; k0 < DMODEL; k0 += 32) {
    v16bf a = load_afrag(Cb + k0, DMODEL, lane);
#pragma unroll
    for (int f = 0; f < 4; ++f)
      acc[f] = wmma_bf16(a,
                         load_bfrag(WUT + (size_t)(n0base + f * 16) * DMODEL + k0, DMODEL, lane),
                         acc[f]);
  }
  const int half = lane >> 4, n = lane & 15;
  float* ob = out + (size_t)rowTile * 16 * DMODEL + n0base;
#pragma unroll
  for (int f = 0; f < 4; ++f)
#pragma unroll
    for (int r = 0; r < 8; ++r)
      ob[(8 * half + r) * DMODEL + f * 16 + n] = acc[f][r];
}

// ---------------- host launch ----------------
extern "C" void kernel_launch(void* const* d_in, const int* in_sizes, int n_in,
                              void* d_out, int out_size, void* d_ws, size_t ws_size,
                              hipStream_t stream) {
  const float* inputs = (const float*)d_in[0];  // [B,T,D]
  const float* WQ     = (const float*)d_in[1];  // [H,D,S]
  const float* WV     = (const float*)d_in[2];  // [H,D,S]
  const float* WU     = (const float*)d_in[3];  // [H*S,D]
  float* out = (float*)d_out;                   // [B,T,D] f32

  char* ws = (char*)d_ws;
  size_t off = 0;
  auto take = [&](size_t bytes) { char* p = ws + off; off = (off + bytes + 255) & ~(size_t)255; return p; };

  const size_t nX  = (size_t)BATCH * TLEN * DMODEL;         // 4,194,304
  const size_t nQV = (size_t)BATCH * NHEADS * TLEN * HSZ;   // 4,194,304
  bf16_t* Xbf  = (bf16_t*)take(nX * 2);
  bf16_t* WQT  = (bf16_t*)take((size_t)NHEADS * DMODEL * HSZ * 2);
  bf16_t* WVT  = (bf16_t*)take((size_t)NHEADS * DMODEL * HSZ * 2);
  bf16_t* WUT  = (bf16_t*)take((size_t)DMODEL * DMODEL * 2);
  bf16_t* Qbf  = (bf16_t*)take(nQV * 2);
  bf16_t* Vbf  = (bf16_t*)take(nQV * 2);
  bf16_t* VTbf = (bf16_t*)take(nQV * 2);
  bf16_t* Cbf  = (bf16_t*)take(nX * 2);

  // prep: convert + transpose weights
  cvt_f32_bf16<<<(int)((nX + 255) / 256), 256, 0, stream>>>(inputs, Xbf, (int)nX);
  {
    int n = NHEADS * DMODEL * HSZ;
    transpose_to_bf16<<<(n + 255) / 256, 256, 0, stream>>>(WQ, WQT, DMODEL, HSZ, n);
    transpose_to_bf16<<<(n + 255) / 256, 256, 0, stream>>>(WV, WVT, DMODEL, HSZ, n);
    int nu = DMODEL * DMODEL;
    transpose_to_bf16<<<(nu + 255) / 256, 256, 0, stream>>>(WU, WUT, DMODEL, DMODEL, nu);
  }

  // projections: 4096 waves (B*H*T/16), 4 waves/block
  proj_kernel<<<1024, 128, 0, stream>>>(Xbf, WQT, WVT, Qbf, Vbf, VTbf);
  // banded attention: 4096 waves
  attn_kernel<<<1024, 128, 0, stream>>>(Qbf, Vbf, VTbf, Cbf);
  // output projection: 4096 waves
  outproj_kernel<<<1024, 128, 0, stream>>>(Cbf, WUT, out);
}
